// MyLSTMLayers_6416681140975
// MI455X (gfx1250) — matrix operand order repaired and checked by reading, not compiled, not run on hardware
//
#include <hip/hip_runtime.h>
#include <hip/hip_bf16.h>

// ---------------------------------------------------------------------------
// 2-layer LSTM, B=128, T=512, D=H=512.  bf16 WMMA GEMM per timestep, fused
// gate activations + state update.
//  - W pre-swizzled once per call into B-fragment layout (global b128 loads,
//    L2-resident, no LDS round trip for B).
//  - A (concat[x_t | h_prev]) staged fp32 -> LDS via async global->LDS copies
//    (double buffered, ASYNCcnt pipeline), converted to bf16 at fragment load.
//  - Native v_tanh_f32 / v_exp_f32 / v_rcp_f32 activations (branch-free).
// ---------------------------------------------------------------------------

typedef __attribute__((ext_vector_type(16))) __bf16 v16bf;
typedef __attribute__((ext_vector_type(8)))  float  v8f;
typedef __attribute__((ext_vector_type(4)))  int    v4i;

#define B_     128
#define T_     512
#define D_     512
#define H_     512
#define L_     2
#define KTOT   1024          // D + H
#define NG     2048          // 4*H
#define BH     (B_ * H_)     // 65536
#define NKC    (KTOT / 32)   // 32 K-chunks
#define PAF    36            // LDS pitch in floats for A tile (144B, 16B aligned)

#if defined(__HIP_DEVICE_COMPILE__) && defined(__gfx1250__) && \
    __has_builtin(__builtin_amdgcn_global_load_async_to_lds_b128)
#define HAS_ASYNC 1
#else
#define HAS_ASYNC 0
#endif

union Frag {
    v16bf  v;
    uint4  q[2];
    __bf16 e[16];
};

__device__ __forceinline__ float fast_sigmoid(float x) {
    // 1 / (1 + 2^(-x*log2(e)))  -> v_exp_f32 + v_rcp_f32, branch free
    float e = __builtin_amdgcn_exp2f(x * -1.442695040888963f);
    return __builtin_amdgcn_rcpf(1.0f + e);
}

__device__ __forceinline__ float fast_tanh(float x) {
#if defined(__HIP_DEVICE_COMPILE__) && __has_builtin(__builtin_amdgcn_tanhf)
    return __builtin_amdgcn_tanhf(x);            // v_tanh_f32 (gfx1250)
#elif defined(__HIP_DEVICE_COMPILE__) && __has_builtin(__builtin_amdgcn_tanh_f32)
    return __builtin_amdgcn_tanh_f32(x);
#else
    // 2*sigmoid(2x) - 1, branch free
    float e = __builtin_amdgcn_exp2f(x * -2.885390081777927f);
    return __builtin_amdgcn_rcpf(1.0f + e) * 2.0f - 1.0f;
#endif
}

__device__ __forceinline__ void wait_async_le(int kcLast) {
#if HAS_ASYNC
    if (kcLast) {
#if __has_builtin(__builtin_amdgcn_s_wait_asynccnt)
        __builtin_amdgcn_s_wait_asynccnt(0);
#else
        asm volatile("s_wait_asynccnt 0" ::: "memory");
#endif
    } else {
#if __has_builtin(__builtin_amdgcn_s_wait_asynccnt)
        __builtin_amdgcn_s_wait_asynccnt(4);
#else
        asm volatile("s_wait_asynccnt 4" ::: "memory");
#endif
    }
#endif
}

// ---------------------------------------------------------------------------
// W [L, 1024, 2048] fp32 -> bf16, swizzled into B-fragment layout:
//   Wsw[l][ntile(128)][kc(32)][lane(32)] = 32 bytes (8 x 2 bf16 K-pairs)
//   ntile = column tile of 16;  lane<16: n=ntile*16+lane, K=kc*32 + 2i,2i+1
//                               lane>=16: same n, K offset +16
// One thread per (l,ntile,kc,lane) writes 32 contiguous bytes (coalesced).
// ---------------------------------------------------------------------------
__global__ __launch_bounds__(256) void swizzle_w_kernel(
    const float* __restrict__ W, uint4* __restrict__ Wsw)
{
    int tidg  = blockIdx.x * 256 + threadIdx.x;   // 0 .. L*128*32*32-1
    int lane  = tidg & 31;
    int kc    = (tidg >> 5) & 31;
    int ntile = (tidg >> 10) & 127;
    int l     = tidg >> 17;
    int n     = ntile * 16 + (lane & 15);
    int kb    = kc * 32 + (lane >> 4) * 16;
    const float* src = W + (size_t)l * KTOT * NG + (size_t)kb * NG + n;
    Frag fr;
    #pragma unroll
    for (int i = 0; i < 8; ++i) {
        fr.e[2 * i]     = (__bf16)src[(size_t)(2 * i) * NG];
        fr.e[2 * i + 1] = (__bf16)src[(size_t)(2 * i + 1) * NG];
    }
    uint4* dst = Wsw + (size_t)tidg * 2;
    dst[0] = fr.q[0];
    dst[1] = fr.q[1];
}

// ---------------------------------------------------------------------------
// Copy initial hiddens [L,2,B,H] into ping-pong h buffers (slot 0) and c buf
// ---------------------------------------------------------------------------
__global__ __launch_bounds__(256) void init_state_kernel(
    const float* __restrict__ hid, float* __restrict__ hbuf,
    float* __restrict__ cbuf)
{
    int i   = blockIdx.x * 256 + threadIdx.x;     // 0 .. L*2*BH-1
    int l   = i >> 17;
    int rem = i & (2 * BH - 1);
    int j   = rem >> 16;                          // 0 = h, 1 = c
    int e   = rem & (BH - 1);
    float v = hid[i];
    if (j == 0) hbuf[(size_t)(l * 2 + 0) * BH + e] = v;
    else        cbuf[(size_t)l * BH + e] = v;
}

// ---------------------------------------------------------------------------
// Copy final h/c into d_out tail:  [L,2,B,H]
// ---------------------------------------------------------------------------
__global__ __launch_bounds__(256) void finalize_kernel(
    const float* __restrict__ hbuf, const float* __restrict__ cbuf,
    float* __restrict__ dst, int finSlot)
{
    int i   = blockIdx.x * 256 + threadIdx.x;
    int l   = i >> 17;
    int rem = i & (2 * BH - 1);
    int j   = rem >> 16;
    int e   = rem & (BH - 1);
    float v = (j == 0) ? hbuf[(size_t)(l * 2 + finSlot) * BH + e]
                       : cbuf[(size_t)l * BH + e];
    dst[i] = v;
}

// ---------------------------------------------------------------------------
// One timestep, one layer.  WG = 256 threads = 8 waves.
// WG tile: 128 rows x 32 hidden units across all 4 gate blocks.
// grid.x = H/32 = 16.
// ---------------------------------------------------------------------------
__global__ __launch_bounds__(256) void lstm_step_kernel(
    const float* __restrict__ A0, long lda0,      // cols 0..511 of A
    const float* __restrict__ A1,                 // h_prev, cols 512..1023
    const uint4* __restrict__ Wl,                 // swizzled bf16 W (this layer)
    const float* __restrict__ bias,               // [2048]
    float*       __restrict__ cst,                // c state (in/out)
    float*       __restrict__ hout,               // h_new
    float*       __restrict__ yout,               // outputs [B,T,H] (layer 1)
    int t, int writeOut)
{
    __shared__ __attribute__((aligned(16))) float Asf[2][B_ * PAF];

    const int tid  = threadIdx.x;
    const int lane = tid & 31;
    const int wave = tid >> 5;
    const int wm   = wave & 3;
    const int wn   = wave >> 2;
    const int half = lane >> 4;
    const int l16  = lane & 15;
    const int u0   = blockIdx.x * 32;
    const int nt2  = blockIdx.x * 2;              // first 16-col tile index

    v8f acc[2][4] = {};                           // 2 M-tiles x 4 gate tiles

    // ---- A-chunk staging: 128 rows x 32 fp32 ----
    const int srow = tid >> 1;
    const int sc0  = (tid & 1) * 16;
    auto stage = [&](int kc, int buf) {
        const float* src = (kc < NKC / 2)
            ? (A0 + (size_t)srow * lda0 + kc * 32 + sc0)
            : (A1 + (size_t)srow * H_ + (kc - NKC / 2) * 32 + sc0);
        float* dst = &Asf[buf][srow * PAF + sc0];
#if HAS_ASYNC
        #pragma unroll
        for (int j = 0; j < 4; ++j)
            __builtin_amdgcn_global_load_async_to_lds_b128(
                (__attribute__((address_space(1))) v4i*)(src + 4 * j),
                (__attribute__((address_space(3))) v4i*)(dst + 4 * j),
                0, 0);
#else
        #pragma unroll
        for (int j = 0; j < 4; ++j)
            *(float4*)(dst + 4 * j) = *(const float4*)(src + 4 * j);
#endif
    };

#if HAS_ASYNC
    stage(0, 0);
    stage(1, 1);
#endif

    for (int kc = 0; kc < NKC; ++kc) {
        int bsel;
#if HAS_ASYNC
        bsel = kc & 1;
        wait_async_le(kc == NKC - 1);   // own current-chunk copies done
        __syncthreads();                // everyone's copies done
#else
        bsel = 0;
        __syncthreads();                // WAR: previous iter's readers done
        stage(kc, 0);
        __syncthreads();
#endif

        // ---- A fragments: fp32 LDS -> bf16 pairs (contiguous k = 16*half + s, matches Wsw) ----
        Frag afr[2];
        #pragma unroll
        for (int mi = 0; mi < 2; ++mi) {
            int m = wm * 32 + mi * 16 + l16;
            const float* ap = &Asf[bsel][m * PAF];
            #pragma unroll
            for (int i = 0; i < 8; ++i) {
                int k = half * 16 + i * 2;
                float2 f = *(const float2*)(ap + k);
                afr[mi].e[2 * i]     = (__bf16)f.x;
                afr[mi].e[2 * i + 1] = (__bf16)f.y;
            }
        }

        // ---- B fragments: direct from swizzled global W (2 x b128) ----
        Frag bfr[4];
        #pragma unroll
        for (int g = 0; g < 4; ++g) {
            int ntile = g * 32 + nt2 + wn;
            const uint4* bp =
                Wl + ((size_t)(ntile * 32 + kc) * 32 + lane) * 2;
            bfr[g].q[0] = bp[0];
            bfr[g].q[1] = bp[1];
            __builtin_prefetch(bp + 64, 0, 0);    // next K-chunk block
        }

        #pragma unroll
        for (int mi = 0; mi < 2; ++mi)
            #pragma unroll
            for (int g = 0; g < 4; ++g)
                acc[mi][g] = __builtin_amdgcn_wmma_f32_16x16x32_bf16(
                    false, afr[mi].v, false, bfr[g].v,
                    (short)0, acc[mi][g], false, false);

#if HAS_ASYNC
        __syncthreads();                // all readers of buf bsel done
        if (kc + 2 < NKC) stage(kc + 2, bsel);
#endif
    }
#if !HAS_ASYNC
    __syncthreads();
#endif

    // ---- fused epilogue: bias + activations + c/h update ----
    const int unit = u0 + wn * 16 + l16;
    const float bi = bias[unit];
    const float bf = bias[H_ + unit];
    const float bg = bias[2 * H_ + unit];
    const float bo = bias[3 * H_ + unit];

    #pragma unroll
    for (int mi = 0; mi < 2; ++mi) {
        #pragma unroll
        for (int r = 0; r < 8; ++r) {
            int row = wm * 32 + mi * 16 + half * 8 + r;   // batch index
            float gi = fast_sigmoid(acc[mi][0][r] + bi);
            float gf = fast_sigmoid(acc[mi][1][r] + bf);
            float gg = fast_tanh   (acc[mi][2][r] + bg);
            float go = fast_sigmoid(acc[mi][3][r] + bo);
            size_t idx = (size_t)row * H_ + unit;
            float cn = gf * cst[idx] + gi * gg;
            cst[idx] = cn;
            float hn = go * fast_tanh(cn);
            hout[idx] = hn;
            if (writeOut)
                yout[(size_t)row * (T_ * H_) + (size_t)t * H_ + unit] = hn;
        }
    }
}

// ---------------------------------------------------------------------------
// Host launcher
// ---------------------------------------------------------------------------
extern "C" void kernel_launch(void* const* d_in, const int* in_sizes, int n_in,
                              void* d_out, int out_size, void* d_ws, size_t ws_size,
                              hipStream_t stream) {
    const float* x    = (const float*)d_in[0];   // [B,T,D]
    const float* hid0 = (const float*)d_in[1];   // [L,2,B,H]
    const float* W    = (const float*)d_in[2];   // [L, D+H, 4H]
    const float* bias = (const float*)d_in[3];   // [L, 4H]
    float* out        = (float*)d_out;           // [B,T,H] ++ [L,2,B,H]

    const size_t wswBytes  = (size_t)L_ * KTOT * NG * 2;            // 8 MB bf16
    const size_t hbufBytes = (size_t)L_ * 2 * BH * sizeof(float);   // 1 MB
    const size_t cbufBytes = (size_t)L_ * BH * sizeof(float);       // 0.5 MB
    if (ws_size < wswBytes + hbufBytes + cbufBytes) return;

    uint4* Wsw  = (uint4*)d_ws;
    float* hbuf = (float*)((char*)d_ws + wswBytes);   // [L][2][B*H] ping-pong
    float* cbuf = hbuf + (size_t)L_ * 2 * BH;         // [L][B*H]

    const int nSw = L_ * 128 * 32 * 32;               // 262144 threads
    swizzle_w_kernel<<<dim3(nSw / 256), dim3(256), 0, stream>>>(W, Wsw);
    init_state_kernel<<<dim3((L_ * 2 * BH) / 256), dim3(256), 0, stream>>>(
        hid0, hbuf, cbuf);

    const size_t wlStride = (size_t)128 * 32 * 32 * 2;   // uint4 per layer

    for (int t = 0; t < T_; ++t) {
        const int rb = t & 1;          // read slot
        const int wb = (t + 1) & 1;    // write slot
        // layer 0: A0 = x[:, t, :], A1 = h0_prev
        lstm_step_kernel<<<dim3(H_ / 32), dim3(256), 0, stream>>>(
            x + (size_t)t * D_, (long)T_ * D_,
            hbuf + (size_t)(0 * 2 + rb) * BH,
            Wsw, bias, cbuf,
            hbuf + (size_t)(0 * 2 + wb) * BH,
            nullptr, t, 0);
        // layer 1: A0 = h0_new, A1 = h1_prev; writes outputs
        lstm_step_kernel<<<dim3(H_ / 32), dim3(256), 0, stream>>>(
            hbuf + (size_t)(0 * 2 + wb) * BH, (long)H_,
            hbuf + (size_t)(1 * 2 + rb) * BH,
            Wsw + wlStride, bias + NG, cbuf + BH,
            hbuf + (size_t)(1 * 2 + wb) * BH,
            out, t, 1);
    }

    finalize_kernel<<<dim3((L_ * 2 * BH) / 256), dim3(256), 0, stream>>>(
        hbuf, cbuf, out + (size_t)B_ * T_ * H_, T_ & 1);
}
